// ModernBertAttention_52398601011310
// MI455X (gfx1250) — matrix operand order repaired and checked by reading, not compiled, not run on hardware
//
#include <hip/hip_runtime.h>
#include <hip/hip_bf16.h>
#include <math.h>

typedef __bf16 bf16;
typedef __attribute__((ext_vector_type(16))) __bf16 v16bf;
typedef __attribute__((ext_vector_type(8)))  __bf16 v8bf;
typedef __attribute__((ext_vector_type(8)))  float  v8f;
typedef __attribute__((ext_vector_type(4)))  unsigned int v4u;
typedef __attribute__((ext_vector_type(8)))  int v8i;
typedef __attribute__((ext_vector_type(4)))  int v4i;

#define HEADS 16
#define HD 64
#define SEQ 2048
#define WIN 128
#define NEGBIG (-1.0e30f)

// ---------------- WMMA helpers (contiguous k order k = half*16 + j, same in A and B) ----------------

__device__ __forceinline__ v8f wmma_bf16(v16bf a, v16bf b, v8f c) {
  return __builtin_amdgcn_wmma_f32_16x16x32_bf16(false, a, false, b, (short)0, c,
                                                 false, false);
}

// 16 contiguous bf16 -> one B-fragment register set (element j -> k = half*16 + j)
__device__ __forceinline__ v16bf load_contig16(const bf16* p) {
  v8bf lo = *(const v8bf*)p;
  v8bf hi = *(const v8bf*)(p + 8);
  v16bf r;
#pragma unroll
  for (int i = 0; i < 8; ++i) { r[i] = lo[i]; r[i + 8] = hi[i]; }
  return r;
}

// A fragment 16x32 (MxK) from row-major tile with leading dim ld.
// lane: m = lane&15, half = lane>>4; elem j: k = half*16 + j
__device__ __forceinline__ v16bf load_a_frag(const bf16* tile, int ld) {
  int lane = threadIdx.x & 31;
  int m = lane & 15, half = lane >> 4;
  const bf16* p = tile + (size_t)m * ld + half * 16;
  v8bf lo = *(const v8bf*)p;
  v8bf hi = *(const v8bf*)(p + 8);
  v16bf r;
#pragma unroll
  for (int i = 0; i < 8; ++i) { r[i] = lo[i]; r[i + 8] = hi[i]; }
  return r;
}

// B fragment 32x16 (KxN) sourced from an N-major ("transposed") tile: elem (k,n) at tileT[n*ld+k]
__device__ __forceinline__ v16bf load_bt_frag(const bf16* tileT, int ld) {
  int lane = threadIdx.x & 31;
  int n = lane & 15, half = lane >> 4;
  return load_contig16(tileT + (size_t)n * ld + half * 16);
}

// ---------------- TDM: 2-D tile DMA global -> LDS (CDNA5 ISA ch.8 D# layout) ----------------
// Loads `rows` rows of `cols` bf16 elements (row stride = row_stride_elems in global)
// into LDS at byte offset lds_off, packed contiguously (row pitch = cols*2 bytes).

__device__ __forceinline__ void tdm_load_b_tile(unsigned lds_off, const bf16* gsrc,
                                                unsigned rows, unsigned cols,
                                                unsigned long long row_stride_elems) {
  unsigned long long ga = (unsigned long long)(size_t)gsrc;
  v4u g0;
  g0[0] = 1u;                                              // count=1, user descriptor
  g0[1] = lds_off;                                         // lds_addr (bytes)
  g0[2] = (unsigned)ga;                                    // global_addr[31:0]
  g0[3] = (unsigned)((ga >> 32) & 0x01ffffffu) | (2u << 30); // global_addr[56:32] | type=2

  unsigned td0 = cols, td1 = rows;                         // tensor dims (OOB bounds)
  unsigned t0 = cols, t1 = rows, t2 = 0;                   // tile dims
  unsigned long long s0 = row_stride_elems, s1 = 0;        // dim0/dim1 strides
  v8i g1;
  g1[0] = (int)(1u << 16);                                 // data_size=1 (2 bytes/elem)
  g1[1] = (int)((td0 & 0xffffu) << 16);                    // tensor_dim0[15:0]
  g1[2] = (int)(((td0 >> 16) & 0xffffu) | ((td1 & 0xffffu) << 16));
  g1[3] = (int)(((td1 >> 16) & 0xffffu) | ((t0 & 0xffffu) << 16));
  g1[4] = (int)((t1 & 0xffffu) | ((t2 & 0xffffu) << 16));
  g1[5] = (int)(s0 & 0xffffffffu);
  g1[6] = (int)(((s0 >> 32) & 0xffffu) | ((s1 & 0xffffu) << 16));
  g1[7] = (int)((s1 >> 16) & 0xffffffffu);

  v4i z4 = {0, 0, 0, 0};
#if __clang_major__ >= 23
  v8i z8 = {};
  __builtin_amdgcn_tensor_load_to_lds(g0, g1, z4, z4, z8, 0);
#else
  __builtin_amdgcn_tensor_load_to_lds(g0, g1, z4, z4, 0);
#endif
}

// ---------------- conversion kernels ----------------

__global__ void cvt_f32_bf16(const float* __restrict__ in, bf16* __restrict__ out, int n) {
  int i = blockIdx.x * blockDim.x + threadIdx.x;
  if (i < n) out[i] = (bf16)in[i];
}

// in [K,N] f32 -> out [N,K] bf16
__global__ void transpose_cvt(const float* __restrict__ in, bf16* __restrict__ out,
                              int K, int N) {
  int i = blockIdx.x * blockDim.x + threadIdx.x;
  if (i < K * N) {
    int n = i % N, k = i / N;
    out[(size_t)n * K + k] = (bf16)in[i];
  }
}

// ---------------- bf16 GEMM: C[M,N] = A[M,K] * Bt[N,K]^T + bias ----------------
// block = 256 (8 waves); block tile 256x64. Wave w owns rows m0..m0+31 (2 A frags),
// B tile (64 rows x 32 k) staged into LDS by the Tensor Data Mover, double-buffered.

__global__ __launch_bounds__(256) void gemm_bf16(const bf16* __restrict__ A,
                                                 const bf16* __restrict__ Bt,
                                                 const float* __restrict__ bias,
                                                 float* __restrict__ C,
                                                 int M, int N, int K) {
  __shared__ bf16 tileB[2][64 * 32];

  int w = threadIdx.x >> 5;
  int lane = threadIdx.x & 31;
  int half = lane >> 4, col = lane & 15;
  int m0 = blockIdx.x * 256 + w * 32;
  int n0 = blockIdx.y * 64;

  v8f acc[2][4] = {};
  const bf16* Arow = A + (size_t)m0 * K;
  const bf16* Brow = Bt + (size_t)n0 * K;

  // prologue: stage first B tile
  if (threadIdx.x < 32)
    tdm_load_b_tile((unsigned)(size_t)&tileB[0][0], Brow, 64, 32,
                    (unsigned long long)K);

  int buf = 0;
  for (int k = 0; k < K; k += 32, buf ^= 1) {
    bool hasNext = (k + 32) < K;
    __syncthreads();  // everyone done reading tileB[buf^1] from previous iter
    if (threadIdx.x < 32) {
      if (hasNext) {
        tdm_load_b_tile((unsigned)(size_t)&tileB[buf ^ 1][0], Brow + (k + 32), 64, 32,
                        (unsigned long long)K);
        __builtin_amdgcn_s_wait_tensorcnt(1);  // current tile done, next in flight
      } else {
        __builtin_amdgcn_s_wait_tensorcnt(0);
      }
    }
    __syncthreads();  // tileB[buf] visible to all waves

    v16bf a0 = load_a_frag(Arow + k, K);
    v16bf a1 = load_a_frag(Arow + (size_t)16 * K + k, K);
    const bf16* bt = &tileB[buf][0];
    v16bf b0 = load_bt_frag(bt + 0 * 16 * 32, 32);
    v16bf b1 = load_bt_frag(bt + 1 * 16 * 32, 32);
    v16bf b2 = load_bt_frag(bt + 2 * 16 * 32, 32);
    v16bf b3 = load_bt_frag(bt + 3 * 16 * 32, 32);
    acc[0][0] = wmma_bf16(a0, b0, acc[0][0]);
    acc[1][0] = wmma_bf16(a1, b0, acc[1][0]);
    acc[0][1] = wmma_bf16(a0, b1, acc[0][1]);
    acc[1][1] = wmma_bf16(a1, b1, acc[1][1]);
    acc[0][2] = wmma_bf16(a0, b2, acc[0][2]);
    acc[1][2] = wmma_bf16(a1, b2, acc[1][2]);
    acc[0][3] = wmma_bf16(a0, b3, acc[0][3]);
    acc[1][3] = wmma_bf16(a1, b3, acc[1][3]);
  }

#pragma unroll
  for (int p = 0; p < 2; ++p) {
#pragma unroll
    for (int f = 0; f < 4; ++f) {
      int n = n0 + f * 16 + col;
      float bv = bias[n];
#pragma unroll
      for (int i = 0; i < 8; ++i) {
        int row = m0 + p * 16 + i + 8 * half;
        C[(size_t)row * N + n] = acc[p][f][i] + bv;
      }
    }
  }
}

// ---------------- RoPE + head split ----------------
// qkv f32 [B,S,3,H,64], cache f32 [256,32,2]
// -> Q,K bf16 [B*H, S, 64] ; Vt bf16 [B*H, 64, S]

__global__ void rope_split(const float* __restrict__ qkv, const float* __restrict__ cache,
                           bf16* __restrict__ Qo, bf16* __restrict__ Ko,
                           bf16* __restrict__ Vt, int B) {
  int idx = blockIdx.x * blockDim.x + threadIdx.x;
  int total = B * SEQ * HEADS * 32;
  if (idx >= total) return;
  int d2 = idx & 31;
  int h = (idx >> 5) & (HEADS - 1);
  int s = (idx >> 9) % SEQ;
  int b = idx / (SEQ * HEADS * 32);

  const float* base = qkv + ((size_t)(b * SEQ + s) * 3) * 1024 + h * 64 + 2 * d2;
  float q0 = base[0],    q1 = base[1];
  float k0 = base[1024], k1 = base[1025];
  float v0 = base[2048], v1 = base[2049];

  int pos = s & 255;
  float c  = cache[(pos * 32 + d2) * 2 + 0];
  float sn = cache[(pos * 32 + d2) * 2 + 1];

  size_t bh = (size_t)(b * HEADS + h);
  size_t qoff = (bh * SEQ + s) * HD + 2 * d2;
  Qo[qoff]     = (bf16)(q0 * c - q1 * sn);
  Qo[qoff + 1] = (bf16)(q1 * c + q0 * sn);
  Ko[qoff]     = (bf16)(k0 * c - k1 * sn);
  Ko[qoff + 1] = (bf16)(k1 * c + k0 * sn);
  Vt[(bh * HD + 2 * d2)     * SEQ + s] = (bf16)v0;
  Vt[(bh * HD + 2 * d2 + 1) * SEQ + s] = (bf16)v1;
}

// ---------------- sliding-window flash attention ----------------
// grid.x = B*H*(S/128); 8 waves/block, wave w owns 16 query rows.

__global__ __launch_bounds__(256) void attn_window(const bf16* __restrict__ Q,
                                                   const bf16* __restrict__ Kc,
                                                   const bf16* __restrict__ Vt,
                                                   bf16* __restrict__ Oattn, int B) {
  const int nqb = SEQ / 128;
  int w = threadIdx.x >> 5;
  int lane = threadIdx.x & 31;
  int half = lane >> 4, col = lane & 15;
  int bh = blockIdx.x / nqb;
  int q0 = (blockIdx.x % nqb) * 128 + w * 16;
  int b = bh / HEADS, h = bh % HEADS;

  __shared__ bf16 plds[8][16][32];

  const bf16* Qp = Q  + ((size_t)bh * SEQ + q0) * HD;
  const bf16* Kp = Kc + (size_t)bh * SEQ * HD;
  const bf16* Vp = Vt + (size_t)bh * HD * SEQ;

  v16bf qf0 = load_a_frag(Qp, HD);        // d = 0..31
  v16bf qf1 = load_a_frag(Qp + 32, HD);   // d = 32..63

  float mrow[8], lrow[8];
  v8f o[4] = {};
#pragma unroll
  for (int i = 0; i < 8; ++i) { mrow[i] = NEGBIG; lrow[i] = 0.f; }

  int kStart = q0 - WIN; if (kStart < 0) kStart = 0; kStart &= ~31;
  int kEnd = q0 + 16 + WIN; if (kEnd > SEQ) kEnd = SEQ;

  for (int ks = kStart; ks < kEnd; ks += 32) {
    // ---- scores: two 16x16 tiles (keys ks..ks+15, ks+16..ks+31) ----
    v8f sc[2];
#pragma unroll
    for (int t = 0; t < 2; ++t) {
      int srow = ks + t * 16 + col;
      int sclmp = srow < SEQ ? srow : SEQ - 1;
      const bf16* kb = Kp + (size_t)sclmp * HD + half * 16;
      v8f s0 = {};
      s0 = wmma_bf16(qf0, load_contig16(kb), s0);
      s0 = wmma_bf16(qf1, load_contig16(kb + 32), s0);
      sc[t] = s0;
    }
    // ---- mask + online softmax ----
#pragma unroll
    for (int i = 0; i < 8; ++i) {
      int qrow = q0 + i + 8 * half;
      int s0c = ks + col, s1c = ks + 16 + col;
      int d0 = qrow - s0c, d1 = qrow - s1c;
      float v0 = (s0c < SEQ && d0 >= -WIN && d0 <= WIN) ? sc[0][i] * 0.125f : NEGBIG;
      float v1 = (s1c < SEQ && d1 >= -WIN && d1 <= WIN) ? sc[1][i] * 0.125f : NEGBIG;
      float t = fmaxf(v0, v1);
#pragma unroll
      for (int msk = 1; msk < 16; msk <<= 1) t = fmaxf(t, __shfl_xor(t, msk, 16));
      float mnew = fmaxf(mrow[i], t);
      float corr = __expf(mrow[i] - mnew);
      mrow[i] = mnew;
      float e0 = __expf(v0 - mnew);
      float e1 = __expf(v1 - mnew);
      float rs = e0 + e1;
#pragma unroll
      for (int msk = 1; msk < 16; msk <<= 1) rs += __shfl_xor(rs, msk, 16);
      lrow[i] = lrow[i] * corr + rs;
#pragma unroll
      for (int f = 0; f < 4; ++f) o[f][i] *= corr;
      int prow = i + 8 * half;
      plds[w][prow][col]      = (bf16)e0;
      plds[w][prow][col + 16] = (bf16)e1;
    }
    // ---- P (16x32) @ V (32x64) ----
    v16bf pa = load_a_frag(&plds[w][0][0], 32);
    int soff = ks + half * 16; if (soff > SEQ - 16) soff = SEQ - 16;
#pragma unroll
    for (int f = 0; f < 4; ++f) {
      v16bf vb = load_contig16(Vp + (size_t)(f * 16 + col) * SEQ + soff);
      o[f] = wmma_bf16(pa, vb, o[f]);
    }
  }

  // ---- epilogue: O / l -> attn bf16 [B, S, H*64] ----
#pragma unroll
  for (int f = 0; f < 4; ++f) {
#pragma unroll
    for (int i = 0; i < 8; ++i) {
      int row = q0 + i + 8 * half;
      float val = o[f][i] / lrow[i];
      Oattn[((size_t)b * SEQ + row) * 1024 + h * 64 + f * 16 + col] = (bf16)val;
    }
  }
}

// ---------------- launch ----------------

extern "C" void kernel_launch(void* const* d_in, const int* in_sizes, int n_in,
                              void* d_out, int out_size, void* d_ws, size_t ws_size,
                              hipStream_t stream) {
  const float* hidden    = (const float*)d_in[0];
  const float* Wqkv      = (const float*)d_in[1];
  const float* Wqkv_bias = (const float*)d_in[2];
  const float* Wo        = (const float*)d_in[3];
  const float* Wo_bias   = (const float*)d_in[4];
  const float* rope      = (const float*)d_in[5];
  float* out = (float*)d_out;

  int tokens = in_sizes[0] / 1024;   // B*S
  int B = tokens / SEQ;

  size_t off = 0;
  auto alloc = [&](size_t bytes) -> void* {
    void* p = (char*)d_ws + off;
    off += (bytes + 255) & ~(size_t)255;
    return p;
  };
  bf16*  Xb     = (bf16*)alloc((size_t)tokens * 1024 * 2);
  bf16*  WqkvT  = (bf16*)alloc((size_t)3072 * 1024 * 2);
  bf16*  WoT    = (bf16*)alloc((size_t)1024 * 1024 * 2);
  float* qkv    = (float*)alloc((size_t)tokens * 3072 * 4);
  bf16*  Qb     = (bf16*)alloc((size_t)B * HEADS * SEQ * HD * 2);
  bf16*  Kb     = (bf16*)alloc((size_t)B * HEADS * SEQ * HD * 2);
  bf16*  Vtb    = (bf16*)alloc((size_t)B * HEADS * SEQ * HD * 2);
  bf16*  attn   = (bf16*)alloc((size_t)tokens * 1024 * 2);

  int nX = tokens * 1024;
  cvt_f32_bf16<<<(nX + 255) / 256, 256, 0, stream>>>(hidden, Xb, nX);
  transpose_cvt<<<(1024 * 3072 + 255) / 256, 256, 0, stream>>>(Wqkv, WqkvT, 1024, 3072);
  transpose_cvt<<<(1024 * 1024 + 255) / 256, 256, 0, stream>>>(Wo, WoT, 1024, 1024);

  // QKV projection: [tokens,1024] x [1024,3072]
  gemm_bf16<<<dim3(tokens / 256, 3072 / 64), 256, 0, stream>>>(
      Xb, WqkvT, Wqkv_bias, qkv, tokens, 3072, 1024);

  int nR = B * SEQ * HEADS * 32;
  rope_split<<<(nR + 255) / 256, 256, 0, stream>>>(qkv, rope, Qb, Kb, Vtb, B);

  attn_window<<<B * HEADS * (SEQ / 128), 256, 0, stream>>>(Qb, Kb, Vtb, attn, B);

  // Output projection: [tokens,1024] x [1024,1024] -> f32 out
  gemm_bf16<<<dim3(tokens / 256, 1024 / 64), 256, 0, stream>>>(
      attn, WoT, Wo_bias, out, tokens, 1024, 1024);
}
